// RNNAttention_20418274525463
// MI455X (gfx1250) — compile-verified, matrix operand order checked by reading, not run on hardware
//
#include <hip/hip_runtime.h>
#include <hip/hip_bf16.h>

// ---------------- problem constants (from reference) ----------------
#define Vv  128
#define Ee  256
#define Hd  1024
#define Bb  32
#define Ss  1024
#define Mtot (Bb * Ss)          // 32768 rows over (b,s)
#define TSTR 40                 // padded LDS row stride (elems) for V^T: 80B, 16B-aligned

typedef __attribute__((ext_vector_type(16))) __bf16 v16bf;
typedef __attribute__((ext_vector_type(8)))  __bf16 v8bf;
typedef __attribute__((ext_vector_type(8)))  float  v8f;
typedef __attribute__((ext_vector_type(4)))  int    v4i;

#if __has_builtin(__builtin_amdgcn_global_load_async_to_lds_b128)
#define ASYNC_LDS 1
#endif

// ---------------- small device helpers ----------------
__device__ __forceinline__ float fast_tanh(float x) {
#if __has_builtin(__builtin_amdgcn_tanhf)
    return __builtin_amdgcn_tanhf(x);        // v_tanh_f32 (gfx1250 TRANS op)
#else
    return tanhf(x);
#endif
}

// 16B global -> LDS copy; async (ASYNCcnt) when available
__device__ __forceinline__ void copy16_g2l(const __bf16* g, __bf16* l) {
#ifdef ASYNC_LDS
    __builtin_amdgcn_global_load_async_to_lds_b128(
        (__attribute__((address_space(1))) v4i*)g,
        (__attribute__((address_space(3))) v4i*)l, 0, 0);
#else
    *(v8bf*)l = *(const v8bf*)g;
#endif
}
__device__ __forceinline__ void async_wait0() {
#ifdef ASYNC_LDS
#if __has_builtin(__builtin_amdgcn_s_wait_asynccnt)
    __builtin_amdgcn_s_wait_asynccnt(0);
#else
    asm volatile("s_wait_asynccnt 0" ::: "memory");
#endif
#endif
}

// ---------------- WMMA helpers (wave32, gfx1250) ----------------
__device__ __forceinline__ v8f wmma_bf16(v16bf a, v16bf b, v8f c) {
    // D(f32 16x16) = A(bf16 16x32) * B(bf16 32x16) + C
    return __builtin_amdgcn_wmma_f32_16x16x32_bf16(
        false, a, false, b, (short)0, c, false, false);
}

// A-fragment (16x32) / B-fragment (32x16) with K CONTIGUOUS for this lane's
// row/column.  `base` points at this lane's row (row/col = lane&15 of tile).
// ISA 7.12.2: lanes 0-15 hold K {0..7,16..23}, lanes 16-31 hold K {8..15,24..31}.
__device__ __forceinline__ v16bf frag_contig(const __bf16* base, int k0) {
    const int lane = threadIdx.x & 31;
    const int half = lane >> 4;
    v8bf lo = *(const v8bf*)(base + k0 + half * 8);
    v8bf hi = *(const v8bf*)(base + k0 + 16 + half * 8);
    v16bf r;
#pragma unroll
    for (int i = 0; i < 8; ++i) { r[i] = lo[i]; r[i + 8] = hi[i]; }
    return r;
}

// ---------------- K0: f32 -> bf16 conversion ----------------
__global__ void cvt_bf16_kernel(const float* __restrict__ src,
                                __bf16* __restrict__ dst, int n) {
    int i = blockIdx.x * blockDim.x + threadIdx.x;
    if (i < n) dst[i] = (__bf16)src[i];
}

// ---------------- K1: Xp = emb[x] @ Wx^T + Wx_b  (M=32768,K=256,N=1024) ----
__global__ void xp_gemm_kernel(const int* __restrict__ x,
                               const __bf16* __restrict__ embB,
                               const __bf16* __restrict__ WxB,
                               const float* __restrict__ Wx_b,
                               float* __restrict__ Xp) {
    const int id   = blockIdx.x * 4 + (threadIdx.x >> 5);
    const int mt   = id >> 6;                                // 0..2047
    const int nt   = id & 63;                                // 0..63
    const int lane = threadIdx.x & 31;
    const int l15  = lane & 15;
    const int half = lane >> 4;

    const int rowid = x[mt * 16 + l15];                      // vocab gather
    const __bf16* arow = embB + (size_t)rowid * Ee;
    const __bf16* brow = WxB + (size_t)(nt * 16 + l15) * Ee;

    v8f c = {};
#pragma unroll
    for (int k0 = 0; k0 < Ee; k0 += 32)
        c = wmma_bf16(frag_contig(arow, k0), frag_contig(brow, k0), c);

    const int col  = nt * 16 + l15;
    const float bias = Wx_b[col];
#pragma unroll
    for (int r = 0; r < 8; ++r) {
        const int m = mt * 16 + r + 8 * half;
        Xp[(size_t)m * Hd + col] = c[r] + bias;
    }
}

// ---------------- K2: one RNN step: H_t = tanh(Xp_t + H_{t-1} @ Wh^T + b) --
__global__ void rnn_step_kernel(const float* __restrict__ Xp,
                                const __bf16* __restrict__ WhB,
                                const float* __restrict__ Wh_b,
                                __bf16* __restrict__ H, int t) {
    const int id   = blockIdx.x * 4 + (threadIdx.x >> 5);   // 0..127
    const int mt   = id >> 6;                                // 0..1  (B tiles)
    const int nt   = id & 63;                                // 0..63 (Hd tiles)
    const int lane = threadIdx.x & 31;
    const int l15  = lane & 15;
    const int half = lane >> 4;
    const int col  = nt * 16 + l15;

    v8f c;
#pragma unroll
    for (int r = 0; r < 8; ++r) {
        const int b = mt * 16 + r + 8 * half;
        c[r] = Xp[((size_t)b * Ss + t) * Hd + col];
    }

    if (t > 0) {
        const __bf16* arow = H + ((size_t)(mt * 16 + l15) * Ss + (t - 1)) * Hd;
        const __bf16* brow = WhB + (size_t)col * Hd;
#pragma unroll 4
        for (int k0 = 0; k0 < Hd; k0 += 32) {
            __builtin_prefetch(brow + k0 + 256, 0, 3);       // global_prefetch_b8
            c = wmma_bf16(frag_contig(arow, k0), frag_contig(brow, k0), c);
        }
    }

    const float bias = Wh_b[col];
#pragma unroll
    for (int r = 0; r < 8; ++r) {
        const int b = mt * 16 + r + 8 * half;
        H[((size_t)b * Ss + t) * Hd + col] = (__bf16)fast_tanh(c[r] + bias);
    }
}

// ---------------- K3: A = H @ Wa^T + Wa_b  (bf16 out) ----------------------
__global__ void a_gemm_kernel(const __bf16* __restrict__ H,
                              const __bf16* __restrict__ WaB,
                              const float* __restrict__ Wa_b,
                              __bf16* __restrict__ Am) {
    const int id   = blockIdx.x * 4 + (threadIdx.x >> 5);
    const int mt   = id >> 6;
    const int nt   = id & 63;
    const int lane = threadIdx.x & 31;
    const int l15  = lane & 15;
    const int half = lane >> 4;

    const __bf16* arow = H + (size_t)(mt * 16 + l15) * Hd;
    const __bf16* brow = WaB + (size_t)(nt * 16 + l15) * Hd;

    v8f c = {};
#pragma unroll 4
    for (int k0 = 0; k0 < Hd; k0 += 32)
        c = wmma_bf16(frag_contig(arow, k0), frag_contig(brow, k0), c);

    const int col  = nt * 16 + l15;
    const float bias = Wa_b[col];
#pragma unroll
    for (int r = 0; r < 8; ++r) {
        const int m = mt * 16 + r + 8 * half;
        Am[(size_t)m * Hd + col] = (__bf16)(c[r] + bias);
    }
}

// ---------------- K4: flash-style causal attention -------------------------
// scores[s,t] = A[b,s].H[b,t]; softmax over keys s<=t; ctx = P-weighted H.
// One block per (batch, 16-query tile); 16 waves; wave w owns 64 H columns.
// V chunks (32 x Hd bf16) are async-staged raw into LDS (pipelined one chunk
// ahead), then transposed into an s-contiguous padded LDS image so the PV
// B-fragments are two 16B ds loads instead of 16 scattered global d16 loads.
__global__ void __launch_bounds__(512)
flash_attn_kernel(const __bf16* __restrict__ H,
                  const __bf16* __restrict__ Am,
                  __bf16* __restrict__ Cx) {
    const int b    = blockIdx.x >> 6;
    const int t0   = (blockIdx.x & 63) * 16;
    const int wave = threadIdx.x >> 5;
    const int lane = threadIdx.x & 31;
    const int l15  = lane & 15;
    const int half = lane >> 4;
    const int h0   = wave * 64;

    __shared__ __align__(16) __bf16 raw[32 * Hd];      // 64 KB, chunk image [s][h]
    __shared__ __align__(16) __bf16 tr[Hd * TSTR];     // 80 KB, transposed [h][s+pad]
    __shared__ __align__(16) float  sc[16][32];
    __shared__ __align__(16) __bf16 pP[16][32];
    __shared__ float rscale[16];
    __shared__ float mst[16], lst[16];

    const int vs   = threadIdx.x >> 4;   // 0..31 : s row this thread stages
    const int vseg = threadIdx.x & 15;   // 0..15 : 64-wide h segment

    auto stage_chunk = [&](int s0) {     // issue 8x16B copies per thread
        const __bf16* g = H + ((size_t)b * Ss + s0 + vs) * Hd + vseg * 64;
        __bf16* l = &raw[vs * Hd + vseg * 64];
#pragma unroll
        for (int it = 0; it < 8; ++it)
            copy16_g2l(g + it * 8, l + it * 8);
    };

    if (threadIdx.x < 16) { mst[threadIdx.x] = -3.0e38f; lst[threadIdx.x] = 0.f; }
    stage_chunk(0);                       // prefetch first V chunk
    __syncthreads();

    const __bf16* Qrow = H + ((size_t)b * Ss + t0 + l15) * Hd;  // A-frag rows
    v8f o0 = {}, o1 = {}, o2 = {}, o3 = {};

    for (int s0 = 0; s0 < t0 + 16; s0 += 32) {
        // --- drain async copies of this chunk, transpose raw -> tr ---
        async_wait0();
        __syncthreads();
#pragma unroll
        for (int it = 0; it < 8; ++it) {
            const int h = vseg * 64 + it * 8;
            v8bf v = *(const v8bf*)&raw[vs * Hd + h];
#pragma unroll
            for (int j = 0; j < 8; ++j) tr[(h + j) * TSTR + vs] = v[j];
        }
        __syncthreads();
        if (s0 + 32 < t0 + 16) stage_chunk(s0 + 32);   // overlap next chunk

        // --- (a) waves 0,1: 16x16 score tiles over keys [s0+16w, +16) ---
        if (wave < 2) {
            const int sc0 = s0 + wave * 16;
            const __bf16* Krow = Am + ((size_t)b * Ss + sc0 + l15) * Hd;
            v8f c = {};
#pragma unroll 4
            for (int k0 = 0; k0 < Hd; k0 += 32)
                c = wmma_bf16(frag_contig(Qrow, k0), frag_contig(Krow, k0), c);
            const int sg = sc0 + l15;                   // global key index
#pragma unroll
            for (int r = 0; r < 8; ++r) {
                const int tg = t0 + r + 8 * half;       // global query index
                sc[r + 8 * half][wave * 16 + l15] = (sg <= tg) ? c[r] : -3.0e38f;
            }
        }
        __syncthreads();

        // --- (b) online softmax update for the 16 query rows ---
        if (threadIdx.x < 16) {
            const int m = threadIdx.x;
            float mx = mst[m];
            const float mold = mx;
#pragma unroll
            for (int j = 0; j < 32; ++j) mx = fmaxf(mx, sc[m][j]);
            const float fac = __expf(mold - mx);
            float sum = 0.f;
#pragma unroll
            for (int j = 0; j < 32; ++j) {
                const float p = __expf(sc[m][j] - mx);
                pP[m][j] = (__bf16)p;
                sum += p;
            }
            rscale[m] = fac;
            mst[m]    = mx;
            lst[m]    = lst[m] * fac + sum;
        }
        __syncthreads();

        // --- (c) all waves: rescale accumulators, P (16x32) x V (32x64) ---
        float s8[8];
#pragma unroll
        for (int r = 0; r < 8; ++r) s8[r] = rscale[r + 8 * half];
#pragma unroll
        for (int r = 0; r < 8; ++r) {
            o0[r] *= s8[r]; o1[r] *= s8[r]; o2[r] *= s8[r]; o3[r] *= s8[r];
        }
        v16bf pf = frag_contig(&pP[l15][0], 0);          // A-frag from LDS
        o0 = wmma_bf16(pf, frag_contig(&tr[(h0 +  0 + l15) * TSTR], 0), o0);
        o1 = wmma_bf16(pf, frag_contig(&tr[(h0 + 16 + l15) * TSTR], 0), o1);
        o2 = wmma_bf16(pf, frag_contig(&tr[(h0 + 32 + l15) * TSTR], 0), o2);
        o3 = wmma_bf16(pf, frag_contig(&tr[(h0 + 48 + l15) * TSTR], 0), o3);
        __syncthreads();                                 // tr/sc/pP reused next iter
    }

    // --- epilogue: normalize by l and store context (bf16) ---
    float inv[8];
#pragma unroll
    for (int r = 0; r < 8; ++r) inv[r] = 1.f / lst[r + 8 * half];
#pragma unroll
    for (int r = 0; r < 8; ++r) {
        const size_t row = (size_t)b * Ss + t0 + r + 8 * half;
        __bf16* out = Cx + row * Hd + h0 + l15;
        out[ 0] = (__bf16)(o0[r] * inv[r]);
        out[16] = (__bf16)(o1[r] * inv[r]);
        out[32] = (__bf16)(o2[r] * inv[r]);
        out[48] = (__bf16)(o3[r] * inv[r]);
    }
}

// ---------------- K5: out = [H | ctx] @ fc^T + fc_b  (M=32768,K=2048,N=128) -
__global__ void out_gemm_kernel(const __bf16* __restrict__ H,
                                const __bf16* __restrict__ Cx,
                                const __bf16* __restrict__ fcB,
                                const float* __restrict__ fc_b,
                                float* __restrict__ out) {
    const int id   = blockIdx.x * 4 + (threadIdx.x >> 5);
    const int mt   = id >> 3;                                // 0..2047
    const int nt   = id & 7;                                 // 0..7
    const int lane = threadIdx.x & 31;
    const int l15  = lane & 15;
    const int half = lane >> 4;

    const __bf16* arowH = H  + (size_t)(mt * 16 + l15) * Hd;
    const __bf16* arowC = Cx + (size_t)(mt * 16 + l15) * Hd;
    const __bf16* brow  = fcB + (size_t)(nt * 16 + l15) * (2 * Hd);

    v8f c = {};
#pragma unroll 4
    for (int k0 = 0; k0 < 2 * Hd; k0 += 32) {
        v16bf a = (k0 < Hd) ? frag_contig(arowH, k0) : frag_contig(arowC, k0 - Hd);
        c = wmma_bf16(a, frag_contig(brow, k0), c);
    }

    const int col  = nt * 16 + l15;
    const float bias = fc_b[col];
#pragma unroll
    for (int r = 0; r < 8; ++r) {
        const int m = mt * 16 + r + 8 * half;
        out[(size_t)m * Vv + col] = c[r] + bias;
    }
}

// ---------------- host-side orchestration ----------------------------------
extern "C" void kernel_launch(void* const* d_in, const int* in_sizes, int n_in,
                              void* d_out, int out_size, void* d_ws, size_t ws_size,
                              hipStream_t stream) {
    const int*   x     = (const int*)  d_in[0];
    const float* emb   = (const float*)d_in[1];
    const float* Wx_w  = (const float*)d_in[2];
    const float* Wx_b  = (const float*)d_in[3];
    const float* Wh_w  = (const float*)d_in[4];
    const float* Wh_b  = (const float*)d_in[5];
    const float* Wa_w  = (const float*)d_in[6];
    const float* Wa_b  = (const float*)d_in[7];
    const float* fc_w  = (const float*)d_in[8];
    const float* fc_b  = (const float*)d_in[9];
    float* out = (float*)d_out;

    // workspace carve-up (all chunks 256B-aligned by construction)
    char* p = (char*)d_ws;
    __bf16* embB = (__bf16*)p; p += (size_t)Vv * Ee * 2;          // 64 KB
    __bf16* WxB  = (__bf16*)p; p += (size_t)Hd * Ee * 2;          // 512 KB
    __bf16* WhB  = (__bf16*)p; p += (size_t)Hd * Hd * 2;          // 2 MB
    __bf16* WaB  = (__bf16*)p; p += (size_t)Hd * Hd * 2;          // 2 MB
    __bf16* fcB  = (__bf16*)p; p += (size_t)Vv * 2 * Hd * 2;      // 512 KB
    float*  Xp   = (float*) p; p += (size_t)Mtot * Hd * 4;        // 128 MB
    __bf16* Hh   = (__bf16*)p; p += (size_t)Mtot * Hd * 2;        // 64 MB
    __bf16* Am   = (__bf16*)p; p += (size_t)Mtot * Hd * 2;        // 64 MB
    __bf16* Cx   = (__bf16*)p;                                    // 64 MB

    auto cvt = [&](const float* s, __bf16* d, int n) {
        cvt_bf16_kernel<<<(n + 255) / 256, 256, 0, stream>>>(s, d, n);
    };
    cvt(emb,  embB, Vv * Ee);
    cvt(Wx_w, WxB,  Hd * Ee);
    cvt(Wh_w, WhB,  Hd * Hd);
    cvt(Wa_w, WaB,  Hd * Hd);
    cvt(fc_w, fcB,  Vv * 2 * Hd);

    // K1: embed + input projection  (2048x64 tiles, 4 waves/block)
    xp_gemm_kernel<<<(2048 * 64) / 4, 128, 0, stream>>>(x, embB, WxB, Wx_b, Xp);

    // K2: sequential RNN, one small launch per timestep (graph-friendly)
    for (int t = 0; t < Ss; ++t)
        rnn_step_kernel<<<32, 128, 0, stream>>>(Xp, WhB, Wh_b, Hh, t);

    // K3: attention key projection A = H Wa^T
    a_gemm_kernel<<<(2048 * 64) / 4, 128, 0, stream>>>(Hh, WaB, Wa_b, Am);

    // K4: causal flash attention -> context
    flash_attn_kernel<<<Bb * (Ss / 16), 512, 0, stream>>>(Hh, Am, Cx);

    // K5: output projection over [H | ctx]
    out_gemm_kernel<<<(2048 * 8) / 4, 128, 0, stream>>>(Hh, Cx, fcB, fc_b, out);
}
